// GraphNet_90735479096003
// MI455X (gfx1250) — compile-verified, matrix operand order checked by reading, not run on hardware
//
#include <hip/hip_runtime.h>
#include <stdint.h>

// ---------------------------------------------------------------------------
// GraphNet fully fused WMMA implementation for gfx1250 — transposed dataflow.
//   D^T = W^T x X^T keeps activations in registers across all 4 MLP layers:
//   the WMMA D layout of layer L is lane-compatible with the B operand layout
//   of layer L+1 (B[kt] = f16(D[2kt] ++ D[2kt+1])), so no LDS round trips.
//   Edge kernel fuses gather + concat + MLP + LayerNorm + residual + scatter.
// ---------------------------------------------------------------------------

typedef _Float16 h16;
typedef _Float16 v16h __attribute__((ext_vector_type(16)));
typedef _Float16 v8h  __attribute__((ext_vector_type(8)));
typedef float    v8f  __attribute__((ext_vector_type(8)));
typedef float    v4f  __attribute__((ext_vector_type(4)));

#define GN_NODES 100000
#define GN_EDGES 3200000
#define GN_LAT   16

// Per-MLP preprocessed weight block: f16 transposed + zero-padded weights,
// contiguous f32 biases / LN params. sizeof == 27584 (multiple of 16).
struct WBlk {
    h16 wt_in[64 * 64];   // [n][k], k padded to 64
    h16 wt_h0[64 * 64];
    h16 wt_h1[64 * 64];
    h16 wt_out[16 * 64];  // [n][k], n padded to 16
    float b_in[64], b_h0[64], b_h1[64], b_out[16], ln_g[16], ln_b[16];
};

enum { M_ENCN = 0, M_ENCE = 1, M_EDGE = 2, M_NODE = 3, M_DEC = 4 };

// ---------------------------------------------------------------------------
// Weight preprocessing: f32 [k][n] row-major -> f16 [n][64] transposed+padded
// ---------------------------------------------------------------------------
__global__ void prep_mlp(const float* w_in, const float* b_in,
                         const float* w_h0, const float* b_h0,
                         const float* w_h1, const float* b_h1,
                         const float* w_out, const float* b_out,
                         const float* ln_g, const float* ln_b,
                         int fin, int fout, WBlk* wb) {
    int tid = blockIdx.x * blockDim.x + threadIdx.x;
    int stride = gridDim.x * blockDim.x;
    for (int i = tid; i < 64 * 64; i += stride) {
        int n = i >> 6, k = i & 63;
        wb->wt_in[i] = (k < fin) ? (h16)w_in[k * 64 + n] : (h16)0.f;
        wb->wt_h0[i] = (h16)w_h0[k * 64 + n];
        wb->wt_h1[i] = (h16)w_h1[k * 64 + n];
    }
    for (int i = tid; i < 16 * 64; i += stride) {
        int n = i >> 6, k = i & 63;
        wb->wt_out[i] = (n < fout) ? (h16)w_out[k * fout + n] : (h16)0.f;
    }
    for (int i = tid; i < 64; i += stride) {
        wb->b_in[i] = b_in[i];
        wb->b_h0[i] = b_h0[i];
        wb->b_h1[i] = b_h1[i];
    }
    for (int i = tid; i < 16; i += stride) {
        wb->b_out[i] = (i < fout) ? b_out[i] : 0.f;
        wb->ln_g[i] = ln_g ? ((i < fout) ? ln_g[i] : 1.f) : 1.f;
        wb->ln_b[i] = ln_b ? ((i < fout) ? ln_b[i] : 0.f) : 0.f;
    }
}

__global__ void zero_f32(float* p, long n) {
    long i = (long)blockIdx.x * blockDim.x + threadIdx.x;
    long st = (long)gridDim.x * blockDim.x;
    for (; i < n; i += st) p[i] = 0.f;
}

// ---------------------------------------------------------------------------
// Fragment helpers
// ---------------------------------------------------------------------------
__device__ __forceinline__ v4f ld4(const float* p) { return *(const v4f*)p; }

// Weight A-fragment: two contiguous 16B runs from an LDS row of 64 halves.
//   element i: k = 32*kt + i + (i>=8 ? 8 : 0) + 8*hh
__device__ __forceinline__ v16h wfrag(const h16* wr, int kt, int hh) {
    const v8h lo = *(const v8h*)(wr + kt * 32 + 8 * hh);
    const v8h hi = *(const v8h*)(wr + kt * 32 + 16 + 8 * hh);
    v16h r;
#pragma unroll
    for (int i = 0; i < 8; ++i) { r[i] = lo[i]; r[i + 8] = hi[i]; }
    return r;
}

// Build an f16 B-fragment from four f32 quads (global-loaded input chunks).
__device__ __forceinline__ v16h cvt_frag(v4f a0, v4f a1, v4f b0, v4f b1) {
    v16h r;
#pragma unroll
    for (int i = 0; i < 4; ++i) {
        r[i]      = (h16)a0[i];
        r[4 + i]  = (h16)a1[i];
        r[8 + i]  = (h16)b0[i];
        r[12 + i] = (h16)b1[i];
    }
    return r;
}

// Lane-local D -> next-layer B conversion: B[kt] = f16(D[2kt] ++ D[2kt+1]).
__device__ __forceinline__ v16h cvt_act(v8f p0, v8f p1) {
    v16h r;
#pragma unroll
    for (int i = 0; i < 8; ++i) { r[i] = (h16)p0[i]; r[i + 8] = (h16)p1[i]; }
    return r;
}

// Hidden layer (transposed): D[ct] = W^T_tile x B, bias + leaky_relu(0.01).
template <int KT>
__device__ __forceinline__ void layerT(const v16h* B, v8f* D, const h16* Wt,
                                       const float* bias, int m, int hh) {
#pragma unroll
    for (int ct = 0; ct < 4; ++ct) {
        const h16* wr = Wt + (ct * 16 + m) * 64;
        v8f acc = {};
#pragma unroll
        for (int kt = 0; kt < KT; ++kt)
            acc = __builtin_amdgcn_wmma_f32_16x16x32_f16(
                false, wfrag(wr, kt, hh), false, B[kt], (short)0, acc, false,
                false);
        const float* bp = bias + ct * 16 + 8 * hh;
        v4f b0 = ld4(bp), b1 = ld4(bp + 4);
#pragma unroll
        for (int i = 0; i < 8; ++i) {
            float x = acc[i] + (i < 4 ? b0[i] : b1[i - 4]);
            acc[i] = fmaxf(x, 0.01f * x);   // leaky relu: v_max
        }
        D[ct] = acc;
    }
}

// Output layer (transposed): 16 out-features x 16 rows, K=64, bias only.
__device__ __forceinline__ v8f layer_outT(const v16h* B, const h16* Wt,
                                          const float* bias, int m, int hh) {
    const h16* wr = Wt + m * 64;
    v8f acc = {};
    acc = __builtin_amdgcn_wmma_f32_16x16x32_f16(false, wfrag(wr, 0, hh),
                                                 false, B[0], (short)0, acc,
                                                 false, false);
    acc = __builtin_amdgcn_wmma_f32_16x16x32_f16(false, wfrag(wr, 1, hh),
                                                 false, B[1], (short)0, acc,
                                                 false, false);
    const float* bp = bias + 8 * hh;
    v4f b0 = ld4(bp), b1 = ld4(bp + 4);
#pragma unroll
    for (int i = 0; i < 8; ++i) acc[i] += (i < 4 ? b0[i] : b1[i - 4]);
    return acc;
}

// ---------------------------------------------------------------------------
// Fused MLP kernel. One wave = 16 rows (lane&15 = row, lane>>4 = feature
// half). 8 waves/block = 128 rows. Activations never leave registers.
// ---------------------------------------------------------------------------
template <int MODE>
__global__ void __launch_bounds__(256)
mlp_kernel(const float* __restrict__ in0, const float* __restrict__ nlat,
           const int* __restrict__ srcI, const int* __restrict__ dstI,
           float* __restrict__ out, float* __restrict__ pe,
           const WBlk* __restrict__ wb, long nrows) {
    __shared__ alignas(16) h16 s_w[3 * 64 * 64 + 16 * 64];  // 26 KB weights
    __shared__ alignas(16) float s_b[240];                  // biases + LN

    const int tid = threadIdx.x;
    {   // cooperative weight block load (hot in L2)
        const uint4* wsrc = (const uint4*)wb;
        uint4* wdst = (uint4*)s_w;
        for (int i = tid; i < 1664; i += 256) wdst[i] = wsrc[i];
        const float* bsrc = (const float*)wb->b_in;
        if (tid < 240) s_b[tid] = bsrc[tid];
    }
    __syncthreads();

    const int wave = tid >> 5, lane = tid & 31;
    const int m = lane & 15, hh = lane >> 4;
    const long tile0 = ((long)blockIdx.x * 8 + wave) * 16;
    const long row = tile0 + m;
    const bool ok = row < nrows;
    const v4f z = {};

    // ---- build layer-1 B fragments (X^T) straight from global memory ----
    v16h B[2];
    int dRow = 0;
    if (MODE == M_EDGE) {
        int s = ok ? srcI[row] : 0;
        dRow = ok ? dstI[row] : 0;
        const float* ep = in0 + row * 16 + 8 * hh;
        const float* sp = nlat + (long)s * 16 + 8 * hh;
        const float* dp = nlat + (long)dRow * 16 + 8 * hh;
        __builtin_prefetch((const void*)(in0 + (tile0 + 2048) * 16), 0, 1);
        v4f e0 = ok ? ld4(ep) : z, e1 = ok ? ld4(ep + 4) : z;
        v4f s0 = ok ? ld4(sp) : z, s1 = ok ? ld4(sp + 4) : z;
        v4f d0 = ok ? ld4(dp) : z, d1 = ok ? ld4(dp + 4) : z;
        B[0] = cvt_frag(e0, e1, s0, s1);   // k 0..15: e | k 16..31: n[src]
        B[1] = cvt_frag(d0, d1, z, z);     // k 32..47: n[dst] | pad
    } else if (MODE == M_NODE) {
        const float* np = in0 + row * 16 + 8 * hh;
        float* pp = pe + row * 16 + 8 * hh;
        v4f a0 = ok ? ld4(np) : z, a1 = ok ? ld4(np + 4) : z;
        v4f p0 = ok ? ld4(pp) : z, p1 = ok ? ld4(pp + 4) : z;
        if (ok) { *(v4f*)pp = z; *(v4f*)(pp + 4) = z; }  // consume + reset
        B[0] = cvt_frag(a0, a1, p0, p1);   // k 0..15: n | k 16..31: pe_sum
    } else if (MODE == M_ENCE) {
        v16h bb = {};
#pragma unroll
        for (int j = 0; j < 8; ++j) {
            int k = 8 * hh + j;
            float v = (ok && k < 5) ? in0[row * 5 + k] : 0.f;
            bb[j] = (h16)v;
        }
        B[0] = bb;
    } else {  // M_ENCN / M_DEC: 16 features, pad to 32
        const float* ip = in0 + row * 16 + 8 * hh;
        v4f a0 = ok ? ld4(ip) : z, a1 = ok ? ld4(ip + 4) : z;
        B[0] = cvt_frag(a0, a1, z, z);
    }

    // ---- 4-layer MLP, all activations in registers ----
    constexpr int KT1 = (MODE == M_EDGE) ? 2 : 1;
    v8f D[4];
    layerT<KT1>(B, D, s_w, s_b, m, hh);
    B[0] = cvt_act(D[0], D[1]);
    B[1] = cvt_act(D[2], D[3]);
    layerT<2>(B, D, s_w + 64 * 64, s_b + 64, m, hh);
    B[0] = cvt_act(D[0], D[1]);
    B[1] = cvt_act(D[2], D[3]);
    layerT<2>(B, D, s_w + 2 * 64 * 64, s_b + 128, m, hh);
    B[0] = cvt_act(D[0], D[1]);
    B[1] = cvt_act(D[2], D[3]);
    v8f o = layer_outT(B, s_w + 3 * 64 * 64, s_b + 192, m, hh);
    // o[i] = out-feature (8*hh + i) of row

    if (MODE == M_DEC) {
        if (ok && hh == 0) {
            float2 t;
            t.x = o[0];
            t.y = o[1];
            *(float2*)(out + row * 2) = t;
        }
    } else {
        // LayerNorm over 16 features: 8 intra-lane + 1 cross-half exchange
        float s = 0.f, q = 0.f;
#pragma unroll
        for (int i = 0; i < 8; ++i) { s += o[i]; q += o[i] * o[i]; }
        s += __shfl_xor(s, 16, 32);
        q += __shfl_xor(q, 16, 32);
        float mu = s * 0.0625f;
        float var = q * 0.0625f - mu * mu;
        float rstd = rsqrtf(var + 1e-5f);
        const float* gp = s_b + 208 + 8 * hh;
        const float* bp = s_b + 224 + 8 * hh;
        v4f g0 = ld4(gp), g1 = ld4(gp + 4);
        v4f be0 = ld4(bp), be1 = ld4(bp + 4);
        v4f y0, y1;
#pragma unroll
        for (int i = 0; i < 4; ++i) {
            y0[i] = (o[i] - mu) * rstd * g0[i] + be0[i];
            y1[i] = (o[4 + i] - mu) * rstd * g1[i] + be1[i];
        }
        if (ok) {
            float* op = out + row * 16 + 8 * hh;
            if (MODE == M_EDGE || MODE == M_NODE) {  // residual (in-place)
                v4f r0 = ld4(op), r1 = ld4(op + 4);
                y0 += r0;
                y1 += r1;
            }
            *(v4f*)op = y0;          // global_store_b128
            *(v4f*)(op + 4) = y1;
            if (MODE == M_EDGE) {    // fused scatter-sum (L2 atomics)
                float* pb = pe + (long)dRow * 16 + 8 * hh;
#pragma unroll
                for (int i = 0; i < 4; ++i) {
                    atomicAdd(pb + i, y0[i]);
                    atomicAdd(pb + 4 + i, y1[i]);
                }
            }
        }
    }
}

// ---------------------------------------------------------------------------
// Host launcher
// ---------------------------------------------------------------------------
extern "C" void kernel_launch(void* const* d_in, const int* in_sizes, int n_in,
                              void* d_out, int out_size, void* d_ws,
                              size_t ws_size, hipStream_t stream) {
    (void)in_sizes; (void)n_in; (void)out_size; (void)ws_size;

    const float* node_f = (const float*)d_in[0];
    const float* edge_f = (const float*)d_in[1];
    const int* src = (const int*)d_in[2];
    const int* dst = (const int*)d_in[3];
    auto F = [&](int i) { return (const float*)d_in[i]; };

    // jax pytree flatten order (dict keys sorted):
    // params: dec(4..11), enc_e(12..21), enc_n(22..31),
    //         proc_e[0..2](32..61), proc_n[0..2](62..91)
    // per normalized MLP: b_h0,b_h1,b_in,b_out,ln_b,ln_g,w_h0,w_h1,w_in,w_out
    // dec (no LN):        b_h0,b_h1,b_in,b_out,w_h0,w_h1,w_in,w_out
    const int DEC = 4, ENCE = 12, ENCN = 22, PE0 = 32, PN0 = 62;

    // workspace carve: e[E*16] f32 | n[N*16] | pe[N*16] | 9x WBlk
    float* e = (float*)d_ws;
    float* nl = e + (size_t)GN_EDGES * GN_LAT;
    float* pe = nl + (size_t)GN_NODES * GN_LAT;
    uintptr_t wbase =
        ((uintptr_t)(pe + (size_t)GN_NODES * GN_LAT) + 255) & ~(uintptr_t)255;
    WBlk* wb = (WBlk*)wbase;

    auto prepN = [&](int base, int fin, int fout, WBlk* w) {
        prep_mlp<<<dim3(16), dim3(256), 0, stream>>>(
            F(base + 8), F(base + 2), F(base + 6), F(base + 0), F(base + 7),
            F(base + 1), F(base + 9), F(base + 3), F(base + 5), F(base + 4),
            fin, fout, w);
    };
    prepN(ENCN, 16, 16, &wb[0]);
    prepN(ENCE, 5, 16, &wb[1]);
    for (int i = 0; i < 3; ++i) prepN(PE0 + 10 * i, 48, 16, &wb[2 + i]);
    for (int i = 0; i < 3; ++i) prepN(PN0 + 10 * i, 32, 16, &wb[5 + i]);
    prep_mlp<<<dim3(16), dim3(256), 0, stream>>>(
        F(DEC + 6), F(DEC + 2), F(DEC + 4), F(DEC + 0), F(DEC + 5), F(DEC + 1),
        F(DEC + 7), F(DEC + 3), nullptr, nullptr, 16, 2, &wb[8]);

    zero_f32<<<dim3(512), dim3(256), 0, stream>>>(pe, (long)GN_NODES * GN_LAT);

    auto grid = [](long rows) { return dim3((unsigned)((rows + 127) / 128)); };

    mlp_kernel<M_ENCN><<<grid(GN_NODES), dim3(256), 0, stream>>>(
        node_f, nullptr, nullptr, nullptr, nl, nullptr, &wb[0], GN_NODES);
    mlp_kernel<M_ENCE><<<grid(GN_EDGES), dim3(256), 0, stream>>>(
        edge_f, nullptr, nullptr, nullptr, e, nullptr, &wb[1], GN_EDGES);

    for (int i = 0; i < 3; ++i) {
        mlp_kernel<M_EDGE><<<grid(GN_EDGES), dim3(256), 0, stream>>>(
            e, nl, src, dst, e, pe, &wb[2 + i], GN_EDGES);
        mlp_kernel<M_NODE><<<grid(GN_NODES), dim3(256), 0, stream>>>(
            nl, nullptr, nullptr, nullptr, nl, pe, &wb[5 + i], GN_NODES);
    }

    mlp_kernel<M_DEC><<<grid(GN_NODES), dim3(256), 0, stream>>>(
        nl, nullptr, nullptr, nullptr, (float*)d_out, nullptr, &wb[8],
        GN_NODES);
}
